// Multi_DefConv_77807627534489
// MI455X (gfx1250) — matrix operand order repaired and checked by reading, not compiled, not run on hardware
//
#include <hip/hip_runtime.h>

// ---------------------------------------------------------------------------
// Multi-image deformable-conv pipeline for MI455X (gfx1250, wave32, WMMA).
// Activations stored HWC as [m][c] (m = y*128 + x), bf16 bits in ushort for
// WMMA A/B operands, f32 accumulators kept in registers across all 9 taps.
// B tiles (and conv A tiles) staged to LDS via global_load_async_to_lds_b128.
// ---------------------------------------------------------------------------

#define HW 128
#define EMB 128
#define NPIX (HW * HW)          // 16384
#define KTAPS 9
#define AP (EMB + 8)            // padded LDS pitch (shorts) to dodge bank conflicts

typedef __attribute__((ext_vector_type(16))) __bf16         v16bf;
typedef __attribute__((ext_vector_type(8)))  float          v8f;
typedef __attribute__((ext_vector_type(8)))  unsigned short v8u;
typedef __attribute__((ext_vector_type(16))) unsigned short v16u;

__device__ __forceinline__ unsigned short f2bf(float f) {
    unsigned u = __float_as_uint(f);
    u += 0x7FFFu + ((u >> 16) & 1u);          // round-to-nearest-even
    return (unsigned short)(u >> 16);
}
__device__ __forceinline__ float bf2f(unsigned short s) {
    return __uint_as_float(((unsigned)s) << 16);
}

// CDNA5 async global->LDS copy (16B), tracked by ASYNCcnt.
__device__ __forceinline__ void async_b128(void* lds, const void* g) {
    unsigned loff = (unsigned)(size_t)lds;    // low 32 bits = LDS byte address
    asm volatile("global_load_async_to_lds_b128 %0, %1, off"
                 :: "v"(loff), "v"(g) : "memory");
}
__device__ __forceinline__ void wait_async0() {
    asm volatile("s_wait_asynccnt 0x0" ::: "memory");
}

// --------------------------- small utility kernels -------------------------

__global__ __launch_bounds__(256) void k_tobf16(unsigned short* __restrict__ dst,
                                                const float* __restrict__ src, int n) {
    int i = blockIdx.x * 256 + threadIdx.x;
    if (i < n) dst[i] = f2bf(src[i]);
}

// repack cw (O,I,3,3) f32 -> Wt[tap][o][i] bf16 (per-tap row-major O x I)
__global__ __launch_bounds__(256) void k_repack(unsigned short* __restrict__ Wt,
                                                const float* __restrict__ cw) {
    int i = blockIdx.x * 256 + threadIdx.x;   // EMB*EMB*9 threads
    if (i >= EMB * EMB * KTAPS) return;
    int tap = i % KTAPS;
    int rem = i / KTAPS;                      // o*EMB + ii
    Wt[tap * (EMB * EMB) + rem] = f2bf(cw[i]);
}

// ------------------------- fused 9-tap WMMA GEMM ---------------------------
// mode 0 (conv):    C[m][n] = cb[n] + sum_tap X[shift_tap(m)][:] . W_tap[n][:]
// mode 1 (deform):  C[m][n] = cb[n] + sum_tap (mask*bilinear(X))[m][:] . W_tap[n][:]
// Block: 256 thr = 8 waves; block owns a 16(M) x 128(N) strip, wave = 1 N-tile.
// Accumulators live in registers across all taps; LDS-staged operands.
__global__ __launch_bounds__(256) void k_gemm9(float* __restrict__ C,
                                               const unsigned short* __restrict__ X,
                                               const unsigned short* __restrict__ W9,
                                               const float* __restrict__ cb,
                                               const float* __restrict__ OM,
                                               int mode) {
    __shared__ unsigned short As[16 * AP];    // 16 x 128 bf16 A tile (padded)
    __shared__ unsigned short Bs[EMB * AP];   // 128 x 128 bf16 W tap (padded)

    const int tid  = threadIdx.x;
    const int lane = tid & 31, wave = tid >> 5;
    const int m0   = blockIdx.x * 16;         // 1024 blocks
    const int n0   = wave * 16;
    const int half = lane >> 4, ln = lane & 15;
    const int cn   = n0 + ln;

    const int y  = m0 >> 7;                   // all 16 tile rows share y
    const int x0 = m0 & (HW - 1);

    // staging role of this thread: row ra (pixel), 8-channel chunk ch
    const int ra = tid >> 4;                  // 0..15
    const int ch = (tid & 15) * 8;            // 0,8,...,120

    v8f c;
    {
        const float bias = cb[cn];
#pragma unroll
        for (int v = 0; v < 8; ++v) c[v] = bias;
    }

    for (int tap = 0; tap < KTAPS; ++tap) {
        // ---- stage B: full 128x128 tap weights -> LDS (async, 16B/copy) ----
        const unsigned short* wt = W9 + (size_t)tap * EMB * EMB;
#pragma unroll
        for (int j = 0; j < 8; ++j) {
            const int e  = tid + j * 256;     // 16B chunk id, 2048 total
            const int o  = e >> 4;            // weight row
            const int c8 = (e & 15) * 8;      // element offset in row
            async_b128(&Bs[o * AP + c8], wt + o * EMB + c8);
        }

        // ---- stage A tile ----
        if (mode == 0) {
            const int dy = tap / 3 - 1, dx = tap % 3 - 1;
            const int sy = y + dy, sx = x0 + ra + dx;
            if ((sy >= 0) & (sy < HW) & (sx >= 0) & (sx < HW)) {
                async_b128(&As[ra * AP + ch],
                           X + ((size_t)(sy * HW + sx)) * EMB + ch);
            } else {
                *(v8u*)(&As[ra * AP + ch]) = (v8u)0;
            }
        } else {
            // deformable bilinear gather straight into the LDS A tile
            const int m = m0 + ra;
            const float* om = OM + (size_t)m * 27;
            const float offy = om[tap], offx = om[9 + tap], msk = om[18 + tap];
            const float py = (float)(m >> 7) - 1.f + (float)(tap / 3) + offy;
            const float px = (float)(m & (HW - 1)) - 1.f + (float)(tap % 3) + offx;
            const float fy0 = floorf(py), fx0 = floorf(px);
            const float wy1 = py - fy0, wx1 = px - fx0;
            const float wy[2] = {1.f - wy1, wy1};
            const float wx[2] = {1.f - wx1, wx1};
            float acc[8];
#pragma unroll
            for (int j = 0; j < 8; ++j) acc[j] = 0.f;
#pragma unroll
            for (int iy = 0; iy < 2; ++iy) {
                const float yy = fy0 + (float)iy;
                if (yy < 0.f || yy > (float)(HW - 1)) continue;
#pragma unroll
                for (int ix = 0; ix < 2; ++ix) {
                    const float xx = fx0 + (float)ix;
                    if (xx < 0.f || xx > (float)(HW - 1)) continue;
                    const float w = wy[iy] * wx[ix];
                    const v8u s = *(const v8u*)(
                        X + ((size_t)((int)yy * HW + (int)xx)) * EMB + ch);
#pragma unroll
                    for (int j = 0; j < 8; ++j)
                        acc[j] = fmaf(w, bf2f(s[j]), acc[j]);
                }
            }
            v8u outv;
#pragma unroll
            for (int j = 0; j < 8; ++j) outv[j] = f2bf(acc[j] * msk);
            *(v8u*)(&As[ra * AP + ch]) = outv;
        }

        wait_async0();
        __syncthreads();

        // prefetch next tap's weights while we compute
        if (tap + 1 < KTAPS)
            __builtin_prefetch(W9 + (size_t)(tap + 1) * EMB * EMB, 0, 0);

        // ---- 4 x WMMA over K=128 ----
        const unsigned short* arow = &As[ln * AP];
        const unsigned short* brow = &Bs[cn * AP];
#pragma unroll
        for (int kk = 0; kk < EMB; kk += 32) {
            v8u a0 = *(const v8u*)(arow + kk + half * 8);
            v8u a1 = *(const v8u*)(arow + kk + 16 + half * 8);
            v8u b0 = *(const v8u*)(brow + kk + half * 8);
            v8u b1 = *(const v8u*)(brow + kk + 16 + half * 8);
            v16u au, bu;
#pragma unroll
            for (int e = 0; e < 8; ++e) {
                au[e] = a0[e]; au[8 + e] = a1[e];
                bu[e] = b0[e]; bu[8 + e] = b1[e];
            }
            c = __builtin_amdgcn_wmma_f32_16x16x32_bf16(
                    false, __builtin_bit_cast(v16bf, au),
                    false, __builtin_bit_cast(v16bf, bu),
                    (short)0, c, false, false);
        }
        __syncthreads();          // LDS reused next tap
    }

    // single store of the accumulated tile
#pragma unroll
    for (int v = 0; v < 8; ++v)
        C[(size_t)(m0 + half * 8 + v) * EMB + cn] = c[v];
}

// -------------------- offsets + softmax(mask) per pixel --------------------
// OM[m][0..8]=off_y, [9..17]=off_x, [18..26]=softmax mask
__global__ __launch_bounds__(256) void k_offmask(float* __restrict__ OM,
                                                 const float* __restrict__ A,
                                                 const float* __restrict__ ow,
                                                 const float* __restrict__ mw) {
    const int m = blockIdx.x * 256 + threadIdx.x;
    const float* row = A + (size_t)m * EMB;
    float accO[18];
    float accM[9];
#pragma unroll
    for (int j = 0; j < 18; ++j) accO[j] = 0.f;
#pragma unroll
    for (int t = 0; t < 9; ++t) accM[t] = 0.f;

    for (int cidx = 0; cidx < EMB; ++cidx) {
        const float v = row[cidx];
#pragma unroll
        for (int j = 0; j < 18; ++j) accO[j] = fmaf(v, ow[j * EMB + cidx], accO[j]);
#pragma unroll
        for (int t = 0; t < 9; ++t) accM[t] = fmaf(v, mw[t * EMB + cidx], accM[t]);
    }
    float mx = accM[0];
#pragma unroll
    for (int t = 1; t < 9; ++t) mx = fmaxf(mx, accM[t]);
    float e[9], s = 0.f;
#pragma unroll
    for (int t = 0; t < 9; ++t) { e[t] = __expf(accM[t] - mx); s += e[t]; }
    const float inv = 1.f / s;

    float* om = OM + (size_t)m * 27;
#pragma unroll
    for (int t = 0; t < 9; ++t) {
        om[t]      = accO[2 * t];       // off_y
        om[9 + t]  = accO[2 * t + 1];   // off_x
        om[18 + t] = e[t] * inv;        // mask
    }
}

// ------------------ per-channel batch-norm + ReLU (N=1) --------------------
__global__ __launch_bounds__(256) void k_bnorm_relu(float* __restrict__ A,
                                                    unsigned short* __restrict__ Xb,
                                                    const float* __restrict__ bg,
                                                    const float* __restrict__ bb) {
    const int o = blockIdx.x;
    const int tid = threadIdx.x;
    __shared__ float r1[256], r2[256];
    float s = 0.f, s2 = 0.f;
    for (int m = tid; m < NPIX; m += 256) {
        float v = A[(size_t)m * EMB + o];
        s += v; s2 += v * v;
    }
    r1[tid] = s; r2[tid] = s2;
    __syncthreads();
    for (int st = 128; st > 0; st >>= 1) {
        if (tid < st) { r1[tid] += r1[tid + st]; r2[tid] += r2[tid + st]; }
        __syncthreads();
    }
    const float mean = r1[0] * (1.f / NPIX);
    const float var  = r2[0] * (1.f / NPIX) - mean * mean;
    const float sc = rsqrtf(var + 1e-5f) * bg[o];
    const float sh = bb[o] - mean * sc;
    for (int m = tid; m < NPIX; m += 256) {
        float v = fmaxf(A[(size_t)m * EMB + o] * sc + sh, 0.f);
        A[(size_t)m * EMB + o]  = v;
        Xb[(size_t)m * EMB + o] = f2bf(v);
    }
}

// ------------------------------ projections --------------------------------
__global__ __launch_bounds__(256) void k_proj(float* __restrict__ out,
                                              const float* __restrict__ xc,
                                              const float* __restrict__ aw,
                                              const float* __restrict__ xp,
                                              const float* __restrict__ bw) {
    const int m = blockIdx.x * 256 + threadIdx.x;
    const int o = blockIdx.y;
    const float* ar = aw + (size_t)o * EMB;
    const float* xr = xc + (size_t)m * EMB;
    float acc = 0.f;
    for (int cidx = 0; cidx < EMB; ++cidx) acc = fmaf(xr[cidx], ar[cidx], acc);
    if (bw) {
        const float* br = bw + (size_t)o * EMB;
        const float* pr = xp + (size_t)m * EMB;
        for (int cidx = 0; cidx < EMB; ++cidx) acc = fmaf(pr[cidx], br[cidx], acc);
    }
    out[(size_t)o * NPIX + m] = acc;
}

// ------------------------- final layer norm over W -------------------------
__global__ __launch_bounds__(128) void k_finalln(float* __restrict__ out,
                                                 const float* __restrict__ lnw,
                                                 const float* __restrict__ lnb) {
    const int row = blockIdx.x;               // 768*128 rows
    const int t = threadIdx.x;
    float v = out[(size_t)row * HW + t];
    __shared__ float r1[128], r2[128];
    r1[t] = v; r2[t] = v * v;
    __syncthreads();
    for (int st = 64; st > 0; st >>= 1) {
        if (t < st) { r1[t] += r1[t + st]; r2[t] += r2[t + st]; }
        __syncthreads();
    }
    const float mu  = r1[0] * (1.f / HW);
    const float var = r2[0] * (1.f / HW) - mu * mu;
    out[(size_t)row * HW + t] = (v - mu) * rsqrtf(var + 1e-5f) * lnw[t] + lnb[t];
}

// ---------------------------------------------------------------------------

extern "C" void kernel_launch(void* const* d_in, const int* in_sizes, int n_in,
                              void* d_out, int out_size, void* d_ws, size_t ws_size,
                              hipStream_t stream) {
    (void)in_sizes; (void)n_in; (void)out_size; (void)ws_size;

    const float* x_in[5];
    for (int i = 0; i < 5; ++i) x_in[i] = (const float*)d_in[i];

    const float* cw[2]  = {(const float*)d_in[5],  (const float*)d_in[11]};
    const float* cbi[2] = {(const float*)d_in[6],  (const float*)d_in[12]};
    const float* ow[2]  = {(const float*)d_in[7],  (const float*)d_in[13]};
    const float* mw[2]  = {(const float*)d_in[8],  (const float*)d_in[14]};
    const float* bg[2]  = {(const float*)d_in[9],  (const float*)d_in[15]};
    const float* bb[2]  = {(const float*)d_in[10], (const float*)d_in[16]};

    const float* paw[5] = {(const float*)d_in[17], (const float*)d_in[18],
                           (const float*)d_in[19], (const float*)d_in[21],
                           (const float*)d_in[23]};
    const float* pbw[5] = {nullptr, nullptr,
                           (const float*)d_in[20], (const float*)d_in[22],
                           (const float*)d_in[24]};
    const int  och[5]  = {20, 80, 150, 210, 308};
    const int  coff[5] = {0, 20, 100, 250, 460};
    const float* lnw = (const float*)d_in[25];
    const float* lnb = (const float*)d_in[26];

    float* out = (float*)d_out;

    // ---- workspace carving ----
    char* p = (char*)d_ws;
    auto carve = [&](size_t bytes) {
        void* r = (void*)p;
        p += (bytes + 255) & ~(size_t)255;
        return r;
    };
    unsigned short* Xb = (unsigned short*)carve((size_t)NPIX * EMB * 2); // block input bf16
    float*          A  = (float*)carve((size_t)NPIX * EMB * 4);          // f32 om / xc
    float*          OM = (float*)carve((size_t)NPIX * 27 * 4);           // off_y/off_x/mask
    unsigned short* Wt = (unsigned short*)carve((size_t)2 * KTAPS * EMB * EMB * 2);

    // ---- repack conv weights (both blocks) ----
    {
        const int n = EMB * EMB * KTAPS;
        const int nb = (n + 255) / 256;
        k_repack<<<nb, 256, 0, stream>>>(Wt, cw[0]);
        k_repack<<<nb, 256, 0, stream>>>(Wt + (size_t)KTAPS * EMB * EMB, cw[1]);
    }

    const int nElem  = NPIX * EMB;
    const int nbElem = nElem / 256;           // 8192
    const int nbGemm = NPIX / 16;             // 1024 (one 16xN strip per block)
    const int nbPix  = NPIX / 256;            // 64

    for (int img = 0; img < 5; ++img) {
        k_tobf16<<<nbElem, 256, 0, stream>>>(Xb, x_in[img], nElem);

        for (int blk = 0; blk < 2; ++blk) {
            const unsigned short* Wblk = Wt + (size_t)blk * KTAPS * EMB * EMB;

            // om = conv3x3(x) + bias (fused 9-tap WMMA GEMM)
            k_gemm9<<<nbGemm, 256, 0, stream>>>(A, Xb, Wblk, cbi[blk], OM, 0);

            // offsets + softmax mask
            k_offmask<<<nbPix, 256, 0, stream>>>(OM, A, ow[blk], mw[blk]);

            // deformable sample + einsum (gather fused into A staging)
            k_gemm9<<<nbGemm, 256, 0, stream>>>(A, Xb, Wblk, cbi[blk], OM, 1);

            // batch-norm + ReLU (A in place, refresh bf16 Xb)
            k_bnorm_relu<<<EMB, 256, 0, stream>>>(A, Xb, bg[blk], bb[blk]);
        }

        dim3 pg(nbPix, och[img]);
        k_proj<<<pg, 256, 0, stream>>>(out + (size_t)coff[img] * NPIX,
                                       A, paw[img], x_in[img], pbw[img]);
    }

    k_finalln<<<768 * HW, 128, 0, stream>>>(out, lnw, lnb);
}
